// AFTLocal_76501957477076
// MI455X (gfx1250) — compile-verified, matrix operand order checked by reading, not run on hardware
//
#include <hip/hip_runtime.h>
#include <hip/hip_bf16.h>

// ---------------------------------------------------------------------------
// AFT-local block: q,k,v = emb@W{q,k,v}+b ; dw[s]=exp(w[s,s]-max_j w[s,j])
// H = sigmoid(q) * (dw*exp(k-max_s k)*v) / (dw*exp(k-max_s k)) ; out = H@Wo+bo
// GEMMs run as bf16 WMMA (v_wmma_f32_16x16x32_bf16), f32 accumulate.
// Wave tile 64x64: 16 WMMA per K-step vs 16 ds_load_b128 (2x the compute
// density of the previous 32x64 version).
// ---------------------------------------------------------------------------

typedef __attribute__((ext_vector_type(16))) __bf16 v16bf;
typedef __attribute__((ext_vector_type(8)))  float  v8f;

static constexpr int BB = 64, SS = 128, DM = 1024, DI = 1024;
static constexpr int MM = BB * SS;           // 8192 rows
static constexpr int BM = 256, BN = 128, BK = 32;
static constexpr int LDT = BK + 8;           // padded LDS stride (elements)

__device__ __forceinline__ unsigned short f2bf(float f) {
    unsigned int u = __float_as_uint(f);
    unsigned int r = u + 0x7FFFu + ((u >> 16) & 1u);   // round-to-nearest-even
    return (unsigned short)(r >> 16);
}

union FragAB { v16bf v; uint4 q[2]; };

// ---------------------------------------------------------------------------
// bf16 WMMA GEMM: C[z] = A(MxK) * B[z]^T-stored(NxK) + bias[z], C is f32 MxN.
// blockIdx.z selects which {B,bias,C} set (fused QKV). 256 thr = 8 waves,
// wave grid 4(M) x 2(N), each wave owns a 64x64 tile (4x4 WMMA accumulators).
// ---------------------------------------------------------------------------
__global__ __launch_bounds__(256)
void wmma_gemm_kernel(const unsigned short* __restrict__ A,
                      const unsigned short* __restrict__ B0,
                      const unsigned short* __restrict__ B1,
                      const unsigned short* __restrict__ B2,
                      const float* __restrict__ bias0,
                      const float* __restrict__ bias1,
                      const float* __restrict__ bias2,
                      float* __restrict__ C0,
                      float* __restrict__ C1,
                      float* __restrict__ C2,
                      int M, int N, int K)
{
    const unsigned short* Bt = (blockIdx.z == 0) ? B0 : (blockIdx.z == 1 ? B1 : B2);
    const float* bias        = (blockIdx.z == 0) ? bias0 : (blockIdx.z == 1 ? bias1 : bias2);
    float* C                 = (blockIdx.z == 0) ? C0 : (blockIdx.z == 1 ? C1 : C2);

    __shared__ __align__(16) unsigned short As[BM * LDT];  // 256x32 bf16 (padded)
    __shared__ __align__(16) unsigned short Bs[BN * LDT];  // 128x32 bf16 (padded)

    const int tid  = threadIdx.x;
    const int lane = tid & 31;
    const int wave = tid >> 5;
    const int r    = lane & 15;
    const int hh   = lane >> 4;            // which 16-lane half of the wave
    const int wm   = (wave & 3) * 64;      // wave tile origin in M (4 waves)
    const int wn   = (wave >> 2) * 64;     // wave tile origin in N (2 waves)
    const int bm0  = blockIdx.y * BM;
    const int bn0  = blockIdx.x * BN;

    // loader mapping: A -> one full 32-elem row per thread (64B contiguous);
    // B -> half row per thread
    const int brow = tid >> 1;
    const int bcol = (tid & 1) << 4;

    v8f c[4][4] = {};

    for (int k0 = 0; k0 < K; k0 += BK) {
        // ---- stage A (BMxBK) and Bt (BNxBK) tiles into LDS ----
        {
            const uint4* ga = reinterpret_cast<const uint4*>(
                A + (size_t)(bm0 + tid) * K + k0);
            uint4 a0 = ga[0], a1 = ga[1], a2 = ga[2], a3 = ga[3];
            uint4* sa = reinterpret_cast<uint4*>(&As[tid * LDT]);
            sa[0] = a0; sa[1] = a1; sa[2] = a2; sa[3] = a3;

            const uint4* gb = reinterpret_cast<const uint4*>(
                Bt + (size_t)(bn0 + brow) * K + k0 + bcol);
            uint4 b0 = gb[0], b1 = gb[1];
            *reinterpret_cast<uint4*>(&Bs[brow * LDT + bcol])     = b0;
            *reinterpret_cast<uint4*>(&Bs[brow * LDT + bcol + 8]) = b1;
        }
        __syncthreads();

        // ---- load WMMA fragments (ISA 16-bit A 16x32 layout) ----
        FragAB fa[4], fb[4];
#pragma unroll
        for (int t = 0; t < 4; ++t) {
            const int row = wm + t * 16 + r;
            fa[t].q[0] = *reinterpret_cast<const uint4*>(&As[row * LDT + hh * 8]);
            fa[t].q[1] = *reinterpret_cast<const uint4*>(&As[row * LDT + 16 + hh * 8]);
        }
#pragma unroll
        for (int u = 0; u < 4; ++u) {
            const int row = wn + u * 16 + r;
            fb[u].q[0] = *reinterpret_cast<const uint4*>(&Bs[row * LDT + hh * 8]);
            fb[u].q[1] = *reinterpret_cast<const uint4*>(&Bs[row * LDT + 16 + hh * 8]);
        }

        // ---- 16 WMMA issues per wave per K-step ----
#pragma unroll
        for (int t = 0; t < 4; ++t)
#pragma unroll
            for (int u = 0; u < 4; ++u)
                c[t][u] = __builtin_amdgcn_wmma_f32_16x16x32_bf16(
                    false, fa[t].v, false, fb[u].v,
                    (short)0, c[t][u], false, false);

        __syncthreads();
    }

    // ---- epilogue: C/D layout VGPR i -> row i+8*half, col lane%16 ----
#pragma unroll
    for (int t = 0; t < 4; ++t) {
#pragma unroll
        for (int u = 0; u < 4; ++u) {
            const int col = bn0 + wn + u * 16 + r;
            const float bv = bias[col];
            v8f acc = c[t][u];
#pragma unroll
            for (int i = 0; i < 8; ++i) {
                const int row = bm0 + wm + t * 16 + hh * 8 + i;
                C[(size_t)row * N + col] = acc[i] + bv;
            }
        }
    }
}

// ---------------------------------------------------------------------------
// helpers
// ---------------------------------------------------------------------------
__global__ void cvt_bf16_kernel(const float* __restrict__ x,
                                unsigned short* __restrict__ y, int n) {
    int i = blockIdx.x * blockDim.x + threadIdx.x;
    if (i < n) y[i] = f2bf(x[i]);
}

// W is (K x N) row-major; Wt becomes (N x K) row-major bf16
__global__ void transcvt_kernel(const float* __restrict__ W,
                                unsigned short* __restrict__ Wt, int K, int N) {
    int i = blockIdx.x * blockDim.x + threadIdx.x;
    if (i < K * N) {
        int k = i % K;
        int n = i / K;
        Wt[i] = f2bf(W[(size_t)k * N + n]);
    }
}

__global__ void dw_kernel(const float* __restrict__ w, float* __restrict__ dw, int S) {
    int i = threadIdx.x;
    if (i < S) {
        float m = -INFINITY;
        for (int j = 0; j < S; ++j) m = fmaxf(m, w[i * S + j]);
        dw[i] = __expf(w[i * S + i] - m);   // diagonal survives the tril mask
    }
}

__global__ void maxk_kernel(const float* __restrict__ Km, float* __restrict__ mk,
                            int S, int D) {
    int d = blockIdx.x * blockDim.x + threadIdx.x;
    int b = blockIdx.y;
    if (d < D) {
        const float* p = Km + (size_t)b * S * D + d;
        float m = -INFINITY;
        for (int s = 0; s < S; ++s) m = fmaxf(m, p[(size_t)s * D]);
        mk[(size_t)b * D + d] = m;
    }
}

__global__ void combine_kernel(const float* __restrict__ Q,
                               const float* __restrict__ Km,
                               const float* __restrict__ V,
                               const float* __restrict__ mk,
                               const float* __restrict__ dw,
                               unsigned short* __restrict__ H,
                               int S, int D, int total) {
    int i = blockIdx.x * blockDim.x + threadIdx.x;
    if (i < total) {
        int d  = i % D;
        int bs = i / D;
        int s  = bs % S;
        int b  = bs / S;
        float q = Q[i], k = Km[i], v = V[i];
        float ek  = __expf(k - mk[(size_t)b * D + d]);
        float dwv = dw[s];
        float num = dwv * (ek * v);
        float den = dwv * ek;
        float sig = 1.0f / (1.0f + __expf(-q));
        H[i] = f2bf(sig * (num / den));
    }
}

// ---------------------------------------------------------------------------
extern "C" void kernel_launch(void* const* d_in, const int* in_sizes, int n_in,
                              void* d_out, int out_size, void* d_ws, size_t ws_size,
                              hipStream_t stream) {
    const float* emb = (const float*)d_in[0];
    const float* Wq  = (const float*)d_in[1];
    const float* bq  = (const float*)d_in[2];
    const float* Wk  = (const float*)d_in[3];
    const float* bk  = (const float*)d_in[4];
    const float* Wv  = (const float*)d_in[5];
    const float* bv  = (const float*)d_in[6];
    const float* w   = (const float*)d_in[7];
    const float* Wo  = (const float*)d_in[8];
    const float* bo  = (const float*)d_in[9];
    float* out = (float*)d_out;

    // workspace carve-up (16B aligned slices)
    char* ws = (char*)d_ws;
    size_t off = 0;
    auto alloc = [&](size_t bytes) { void* p = ws + off; off += (bytes + 15) & ~size_t(15); return p; };
    unsigned short* Xbf = (unsigned short*)alloc((size_t)MM * DM * 2);
    unsigned short* WqT = (unsigned short*)alloc((size_t)DM * DI * 2);
    unsigned short* WkT = (unsigned short*)alloc((size_t)DM * DI * 2);
    unsigned short* WvT = (unsigned short*)alloc((size_t)DM * DI * 2);
    unsigned short* WoT = (unsigned short*)alloc((size_t)DI * DM * 2);
    float* Qb = (float*)alloc((size_t)MM * DI * 4);
    float* Kb = (float*)alloc((size_t)MM * DI * 4);
    float* Vb = (float*)alloc((size_t)MM * DI * 4);
    float* mk = (float*)alloc((size_t)BB * DI * 4);
    float* dw = (float*)alloc((size_t)SS * 4);
    unsigned short* Hbf = (unsigned short*)alloc((size_t)MM * DI * 2);
    (void)ws_size; (void)n_in; (void)in_sizes; (void)out_size;

    const int threads = 256;

    // 1) precision conversion / weight transpose
    cvt_bf16_kernel<<<(MM * DM + threads - 1) / threads, threads, 0, stream>>>(emb, Xbf, MM * DM);
    transcvt_kernel<<<(DM * DI + threads - 1) / threads, threads, 0, stream>>>(Wq, WqT, DM, DI);
    transcvt_kernel<<<(DM * DI + threads - 1) / threads, threads, 0, stream>>>(Wk, WkT, DM, DI);
    transcvt_kernel<<<(DM * DI + threads - 1) / threads, threads, 0, stream>>>(Wv, WvT, DM, DI);
    transcvt_kernel<<<(DI * DM + threads - 1) / threads, threads, 0, stream>>>(Wo, WoT, DI, DM);
    dw_kernel<<<1, SS, 0, stream>>>(w, dw, SS);

    // 2) fused QKV GEMM (z selects weight set)
    dim3 grid1(DI / BN, MM / BM, 3);
    wmma_gemm_kernel<<<grid1, threads, 0, stream>>>(
        Xbf, WqT, WkT, WvT, bq, bk, bv, Qb, Kb, Vb, MM, DI, DM);

    // 3) column-max of K over sequence, then fused elementwise -> H (bf16)
    dim3 gmax(DI / threads, BB, 1);
    maxk_kernel<<<gmax, threads, 0, stream>>>(Kb, mk, SS, DI);
    combine_kernel<<<(MM * DI + threads - 1) / threads, threads, 0, stream>>>(
        Qb, Kb, Vb, mk, dw, Hbf, SS, DI, MM * DI);

    // 4) output GEMM: out = H @ Wo + bo
    dim3 grid2(DM / BN, MM / BM, 1);
    wmma_gemm_kernel<<<grid2, threads, 0, stream>>>(
        Hbf, WoT, WoT, WoT, bo, bo, bo, out, out, out, MM, DM, DI);
}
